// TransformerBlock_22617297780972
// MI455X (gfx1250) — matrix operand order repaired and checked by reading, not compiled, not run on hardware
//
#include <hip/hip_runtime.h>
#include <hip/hip_bf16.h>
#include <math.h>

// ---------------------------------------------------------------------------
// Types for WMMA fragments (v_wmma_f32_16x16x32_f16)
// ---------------------------------------------------------------------------
typedef __attribute__((ext_vector_type(16))) _Float16 f16x16;
typedef __attribute__((ext_vector_type(8)))  _Float16 f16x8;
typedef __attribute__((ext_vector_type(8)))  float    f32x8;

union Frag16 { f16x16 v; f16x8 h[2]; };

#define WMMA_F16(a, b, c)                                                     \
  __builtin_amdgcn_wmma_f32_16x16x32_f16(false, (a), false, (b), (short)0,    \
                                         (c), false, false)

// ---------------------------------------------------------------------------
// Cross-lane reductions via ds_swizzle (group-of-32 XOR mode).
// Masks 1,2,4,8 stay within each 16-lane half (matches WMMA D-layout halves).
// ---------------------------------------------------------------------------
#define SWZ_XOR(m) (((m) << 10) | 0x1F)

template <int IMM>
__device__ __forceinline__ float swz(float x) {
  return __builtin_bit_cast(
      float, __builtin_amdgcn_ds_swizzle(__builtin_bit_cast(int, x), IMM));
}
__device__ __forceinline__ float redMax16(float x) {
  x = fmaxf(x, swz<SWZ_XOR(1)>(x));
  x = fmaxf(x, swz<SWZ_XOR(2)>(x));
  x = fmaxf(x, swz<SWZ_XOR(4)>(x));
  x = fmaxf(x, swz<SWZ_XOR(8)>(x));
  return x;
}
__device__ __forceinline__ float redAdd16(float x) {
  x += swz<SWZ_XOR(1)>(x);
  x += swz<SWZ_XOR(2)>(x);
  x += swz<SWZ_XOR(4)>(x);
  x += swz<SWZ_XOR(8)>(x);
  return x;
}
__device__ __forceinline__ float redAdd32(float x) {
  x = redAdd16(x);
  x += swz<SWZ_XOR(16)>(x);
  return x;
}

__device__ __forceinline__ float geluf(float x) {
  return 0.5f * x * (1.0f + erff(x * 0.70710678118654752f));
}

// ---------------------------------------------------------------------------
// LayerNorm over C=512, one wave (32 lanes) per token, f16 output.
// mapMode==1: token t reads source row  b*Nrow + 1 + (t % NP)   (skip cls)
// ---------------------------------------------------------------------------
__global__ __launch_bounds__(256) void ln_f16_kernel(
    const float* __restrict__ X, const float* __restrict__ G,
    const float* __restrict__ Bv, _Float16* __restrict__ Y, int tokens,
    int mapMode, int NP, int Nrow) {
  const int lane = threadIdx.x & 31;
  const int wave = threadIdx.x >> 5;
  const int t = blockIdx.x * 8 + wave;
  if (t >= tokens) return;  // wave-uniform exit
  int src = t;
  if (mapMode) {
    const int b = t / NP;
    src = b * Nrow + 1 + (t - b * NP);
  }
  const float* x = X + (size_t)src * 512;
  float v[16];
  float s = 0.f;
#pragma unroll
  for (int i = 0; i < 16; ++i) {
    v[i] = x[lane + 32 * i];
    s += v[i];
  }
  s = redAdd32(s);
  const float mean = s * (1.0f / 512.0f);
  float q = 0.f;
#pragma unroll
  for (int i = 0; i < 16; ++i) {
    const float d = v[i] - mean;
    q += d * d;
  }
  q = redAdd32(q);
  const float rstd = rsqrtf(q * (1.0f / 512.0f) + 1e-5f);
  _Float16* y = Y + (size_t)t * 512;
#pragma unroll
  for (int i = 0; i < 16; ++i) {
    const int c = lane + 32 * i;
    y[c] = (_Float16)((v[i] - mean) * rstd * G[c] + Bv[c]);
  }
}

// ---------------------------------------------------------------------------
// Weight convert: W f32 [K][Nout] row-major  ->  Wt f16 [Nout][K] (transposed)
// ---------------------------------------------------------------------------
__global__ void wconv_kernel(const float* __restrict__ W,
                             _Float16* __restrict__ Wt, int K, int Nout) {
  const size_t id = (size_t)blockIdx.x * 256 + threadIdx.x;
  if (id >= (size_t)K * Nout) return;
  const int k = (int)(id % K);
  const int n = (int)(id / K);
  Wt[id] = (_Float16)W[(size_t)k * Nout + n];
}

// ---------------------------------------------------------------------------
// Split fused QKV [tok][1536] into per-head:
//   Q,K : [bh][St][64]  (row-major, hd contiguous -> direct A / B frags)
//   Vt  : [bh][64][St]  (transposed -> direct B frags for P@V)
// Pads (p in [S,St)) are zero-filled.
// ---------------------------------------------------------------------------
__global__ void qkv_split_kernel(const _Float16* __restrict__ qkv,
                                 _Float16* __restrict__ Q,
                                 _Float16* __restrict__ Kd,
                                 _Float16* __restrict__ Vt, int Bseq, int NH,
                                 int S, int St) {
  const size_t id = (size_t)blockIdx.x * 256 + threadIdx.x;
  const size_t total = (size_t)Bseq * NH * St * 64;
  if (id >= total) return;
  const int d = (int)(id & 63);
  size_t r = id >> 6;
  const int p = (int)(r % St);
  r /= St;
  const int h = (int)(r % NH);
  const int s = (int)(r / NH);
  _Float16 qv = (_Float16)0.f, kv = (_Float16)0.f, vv = (_Float16)0.f;
  if (p < S) {
    const size_t base = ((size_t)s * S + p) * 1536 + h * 64 + d;
    qv = qkv[base];
    kv = qkv[base + 512];
    vv = qkv[base + 1024];
  }
  const size_t bh = (size_t)s * NH + h;
  Q[(bh * St + p) * 64 + d] = qv;
  Kd[(bh * St + p) * 64 + d] = kv;
  Vt[(bh * 64 + d) * (size_t)St + p] = vv;
}

// ---------------------------------------------------------------------------
// Generic WMMA GEMM:  out[M][Nout] = A[M][K](f16) @ Wt[Nout][K]^T + bias
// Block: 128 threads = 4 waves; block tile 128(M) x 64(N);
// each wave: 2 m-tiles x 4 n-tiles = 8 WMMAs per 32-wide k-step.
// Register double-buffered k-loop: next chunk's fragments are issued before
// the current chunk's WMMAs execute (partial loadcnt waits, latency hiding).
// mode 0: f16 out              mode 1: f16 out with exact GELU
// mode 2: f32 residual +=      mode 3: f32 residual += with patch row map
// ---------------------------------------------------------------------------
__global__ __launch_bounds__(128) void gemm_wmma_kernel(
    const _Float16* __restrict__ A, const _Float16* __restrict__ Wt,
    const float* __restrict__ bias, int M, int K, int Nout, int mode,
    _Float16* __restrict__ out16, float* __restrict__ outf, int NP, int Nrow) {
  const int lane = threadIdx.x & 31;
  const int wave = threadIdx.x >> 5;  // 0..3
  const int l16 = lane & 15;          // row (A) / col (B,D) within tile
  const int hh = lane >> 4;           // lane half
  const int tm = blockIdx.x * 128 + wave * 32;
  const int tn = blockIdx.y * 64;

  int r0 = tm + l16;
  if (r0 > M - 1) r0 = M - 1;  // clamp pad rows (stores are guarded)
  int r1 = tm + 16 + l16;
  if (r1 > M - 1) r1 = M - 1;
  const _Float16* a0p = A + (size_t)r0 * K;
  const _Float16* a1p = A + (size_t)r1 * K;

  f32x8 acc[2][4];
#pragma unroll
  for (int mi = 0; mi < 2; ++mi)
#pragma unroll
    for (int j = 0; j < 4; ++j)
#pragma unroll
      for (int r = 0; r < 8; ++r) acc[mi][j][r] = 0.0f;

  auto loadA = [&](const _Float16* p, int kc, Frag16& f) {
    f.h[0] = *(const f16x8*)(p + kc + 16 * hh);
    f.h[1] = *(const f16x8*)(p + kc + 16 * hh + 8);
  };
  auto loadB = [&](int j, int kc, Frag16& f) {
    const _Float16* p = Wt + (size_t)(tn + j * 16 + l16) * K + kc + 16 * hh;
    f.h[0] = *(const f16x8*)(p);
    f.h[1] = *(const f16x8*)(p + 8);
  };

  Frag16 a0, a1, b[4];
  loadA(a0p, 0, a0);
  loadA(a1p, 0, a1);
#pragma unroll
  for (int j = 0; j < 4; ++j) loadB(j, 0, b[j]);

  for (int kc = 0; kc < K; kc += 32) {
    // Prefetch next chunk (clamped to 0 on last iter: branch-free, harmless).
    const int kn = (kc + 32 < K) ? (kc + 32) : 0;
    Frag16 a0n, a1n, bn[4];
    loadA(a0p, kn, a0n);
    loadA(a1p, kn, a1n);
#pragma unroll
    for (int j = 0; j < 4; ++j) loadB(j, kn, bn[j]);

#pragma unroll
    for (int j = 0; j < 4; ++j) {
      acc[0][j] = WMMA_F16(a0.v, b[j].v, acc[0][j]);
      acc[1][j] = WMMA_F16(a1.v, b[j].v, acc[1][j]);
    }
    a0 = a0n;
    a1 = a1n;
#pragma unroll
    for (int j = 0; j < 4; ++j) b[j] = bn[j];
  }

#pragma unroll
  for (int mi = 0; mi < 2; ++mi) {
#pragma unroll
    for (int j = 0; j < 4; ++j) {
      const int col = tn + j * 16 + l16;
      const float bv = bias[col];
#pragma unroll
      for (int r = 0; r < 8; ++r) {
        const int rowt = tm + mi * 16 + r + 8 * hh;
        if (rowt < M) {
          float v = acc[mi][j][r] + bv;
          if (mode == 0) {
            out16[(size_t)rowt * Nout + col] = (_Float16)v;
          } else if (mode == 1) {
            out16[(size_t)rowt * Nout + col] = (_Float16)geluf(v);
          } else if (mode == 2) {
            outf[(size_t)rowt * Nout + col] += v;
          } else {
            const int b2 = rowt / NP;
            const int rem = rowt - b2 * NP;
            outf[((size_t)b2 * Nrow + 1 + rem) * Nout + col] += v;
          }
        }
      }
    }
  }
}

// ---------------------------------------------------------------------------
// Flash attention (hd = 64, C = 512). Block = 128 threads = 4 waves.
// Each wave owns 16 query rows; streams keys 32 at a time with online softmax.
// Next key chunk's K fragments are prefetched during softmax; V fragments are
// issued before the softmax VALU block so their latency overlaps exp/reduce.
// grid = (ceil(St/64), NH, Bseq).  O: [seq*S + p][512] f16, heads concat.
// ---------------------------------------------------------------------------
__global__ __launch_bounds__(128) void attn_wmma_kernel(
    const _Float16* __restrict__ Qh, const _Float16* __restrict__ Kh,
    const _Float16* __restrict__ Vt, _Float16* __restrict__ O, int S, int St,
    float scale) {
  __shared__ __align__(16) _Float16 Pb[4][16][32];

  const int lane = threadIdx.x & 31;
  const int wave = threadIdx.x >> 5;
  const int nl = lane & 15;
  const int hh = lane >> 4;
  const int head = blockIdx.y;
  const int NH = gridDim.y;
  const int seq = blockIdx.z;
  const int bh = seq * NH + head;
  const int qt = blockIdx.x * 4 + wave;
  if (qt * 16 >= St) return;  // wave-uniform
  const int q0 = qt * 16;

  const size_t qkBase = (size_t)bh * St * 64;

  // Q A-fragments (16 x 64 = two 16x32 chunks), resident for whole kernel.
  const _Float16* qrow = Qh + qkBase + (size_t)(q0 + nl) * 64;
  Frag16 qa0, qa1;
  qa0.h[0] = *(const f16x8*)(qrow + 16 * hh);
  qa0.h[1] = *(const f16x8*)(qrow + 16 * hh + 8);
  qa1.h[0] = *(const f16x8*)(qrow + 32 + 16 * hh);
  qa1.h[1] = *(const f16x8*)(qrow + 32 + 16 * hh + 8);

  float mrun[8], lrun[8];
  f32x8 acc[4];
#pragma unroll
  for (int d = 0; d < 4; ++d)
#pragma unroll
    for (int r = 0; r < 8; ++r) acc[d][r] = 0.0f;
#pragma unroll
  for (int r = 0; r < 8; ++r) {
    mrun[r] = -1e30f;
    lrun[r] = 0.0f;
  }

  auto loadK = [&](int j0, Frag16 (&kb)[2][2]) {
#pragma unroll
    for (int t = 0; t < 2; ++t) {
      const _Float16* krow = Kh + qkBase + (size_t)(j0 + t * 16 + nl) * 64;
      kb[t][0].h[0] = *(const f16x8*)(krow + 16 * hh);
      kb[t][0].h[1] = *(const f16x8*)(krow + 16 * hh + 8);
      kb[t][1].h[0] = *(const f16x8*)(krow + 32 + 16 * hh);
      kb[t][1].h[1] = *(const f16x8*)(krow + 32 + 16 * hh + 8);
    }
  };

  Frag16 kb[2][2];
  loadK(0, kb);

  for (int j0 = 0; j0 < St; j0 += 32) {
    // --- scores: S = Q @ K^T for 32 keys (two 16-key tiles) ---
    f32x8 sc[2];
#pragma unroll
    for (int t = 0; t < 2; ++t) {
      f32x8 s;
#pragma unroll
      for (int r = 0; r < 8; ++r) s[r] = 0.0f;
      s = WMMA_F16(qa0.v, kb[t][0].v, s);
      s = WMMA_F16(qa1.v, kb[t][1].v, s);
      sc[t] = s;
    }

    // Prefetch next chunk's K fragments (clamped: branch-free).
    const int jn = (j0 + 32 < St) ? (j0 + 32) : 0;
    Frag16 kbn[2][2];
    loadK(jn, kbn);

    // Issue V fragment loads now; latency overlaps softmax VALU below.
    Frag16 vb[4];
#pragma unroll
    for (int d = 0; d < 4; ++d) {
      const _Float16* vrow =
          Vt + ((size_t)bh * 64 + d * 16 + nl) * St + j0 + 16 * hh;
      vb[d].h[0] = *(const f16x8*)(vrow);
      vb[d].h[1] = *(const f16x8*)(vrow + 8);
    }

    // --- scale + mask + row max ---
    float rmax[2][8];
#pragma unroll
    for (int t = 0; t < 2; ++t) {
      const bool colValid = (j0 + t * 16 + nl) < S;
#pragma unroll
      for (int r = 0; r < 8; ++r) {
        float x = sc[t][r] * scale;
        x = colValid ? x : -1e30f;
        sc[t][r] = x;
        rmax[t][r] = redMax16(x);
      }
    }

    // --- online softmax update + stage P into LDS ---
#pragma unroll
    for (int r = 0; r < 8; ++r) {
      const float mnew = fmaxf(mrun[r], fmaxf(rmax[0][r], rmax[1][r]));
      const float alpha = __expf(mrun[r] - mnew);
      mrun[r] = mnew;
      const float p0 = __expf(sc[0][r] - mnew);
      const float p1 = __expf(sc[1][r] - mnew);
      lrun[r] = lrun[r] * alpha + redAdd16(p0 + p1);
#pragma unroll
      for (int d = 0; d < 4; ++d) acc[d][r] *= alpha;
      Pb[wave][r + 8 * hh][nl] = (_Float16)p0;
      Pb[wave][r + 8 * hh][16 + nl] = (_Float16)p1;
    }

    // Drain LDS stores before reloading as A-fragment (in-order per wave).
    asm volatile("s_wait_dscnt 0x0" ::: "memory");

    Frag16 pa;
    const _Float16* prow = &Pb[wave][nl][0];
    pa.h[0] = *(const f16x8*)(prow + 16 * hh);
    pa.h[1] = *(const f16x8*)(prow + 16 * hh + 8);

    // --- O += P @ V ---
#pragma unroll
    for (int d = 0; d < 4; ++d) acc[d] = WMMA_F16(pa.v, vb[d].v, acc[d]);

#pragma unroll
    for (int t = 0; t < 2; ++t) {
      kb[t][0] = kbn[t][0];
      kb[t][1] = kbn[t][1];
    }
  }

  // Epilogue: normalize and store O (heads concatenated along C).
#pragma unroll
  for (int r = 0; r < 8; ++r) {
    const int p = q0 + r + 8 * hh;
    if (p < S) {
      const float inv = 1.0f / lrun[r];
#pragma unroll
      for (int d = 0; d < 4; ++d) {
        O[((size_t)seq * S + p) * 512 + head * 64 + d * 16 + nl] =
            (_Float16)(acc[d][r] * inv);
      }
    }
  }
}

// ---------------------------------------------------------------------------
// Host orchestration
// ---------------------------------------------------------------------------
extern "C" void kernel_launch(void* const* d_in, const int* in_sizes, int n_in,
                              void* d_out, int out_size, void* d_ws,
                              size_t ws_size, hipStream_t stream) {
  (void)in_sizes; (void)n_in; (void)out_size; (void)ws_size;

  constexpr int C = 512, H = 2048, NH = 8, B = 2, P = 196, NI = 16;
  constexpr int N = 1 + NI * P;              // 3137
  constexpr int TG = B * N;                  // 6274 tokens (global)
  constexpr int TL = B * NI * P;             // 6272 tokens (local)
  constexpr int StG = ((N + 31) / 32) * 32;  // 3168
  constexpr int StL = ((P + 31) / 32) * 32;  // 224
  constexpr int BHG = B * NH;                // 16
  constexpr int BHL = B * NI * NH;           // 256
  const size_t qkElems = ((size_t)BHG * StG * 64 > (size_t)BHL * StL * 64)
                             ? (size_t)BHG * StG * 64
                             : (size_t)BHL * StL * 64;

  const float* x_in   = (const float*)d_in[0];
  const float* ln1g   = (const float*)d_in[2];
  const float* ln1b   = (const float*)d_in[3];
  const float* ln2g   = (const float*)d_in[4];
  const float* ln2b   = (const float*)d_in[5];
  const float* ln3g   = (const float*)d_in[6];
  const float* ln3b   = (const float*)d_in[7];
  const float* gqkvw  = (const float*)d_in[8];
  const float* gqkvb  = (const float*)d_in[9];
  const float* gprojw = (const float*)d_in[10];
  const float* gprojb = (const float*)d_in[11];
  const float* lqkvw  = (const float*)d_in[12];
  const float* lqkvb  = (const float*)d_in[13];
  const float* lprojw = (const float*)d_in[14];
  const float* lprojb = (const float*)d_in[15];
  const float* w1     = (const float*)d_in[16];
  const float* b1     = (const float*)d_in[17];
  const float* w2     = (const float*)d_in[18];
  const float* b2     = (const float*)d_in[19];

  float* xf = (float*)d_out;  // f32 residual stream lives in d_out

  // Workspace layout (f16 elements, 256B-aligned chunks)
  _Float16* base = (_Float16*)d_ws;
  size_t off = 0;
  auto alloc = [&](size_t n) {
    _Float16* p = base + off;
    off += (n + 127) & ~(size_t)127;
    return p;
  };
  _Float16* h16   = alloc((size_t)TG * C);
  _Float16* qkv16 = alloc((size_t)TG * 3 * C);
  _Float16* Qb    = alloc(qkElems);
  _Float16* Kb    = alloc(qkElems);
  _Float16* Vb    = alloc(qkElems);
  _Float16* O16   = alloc((size_t)TG * C);
  _Float16* g16   = alloc((size_t)TG * H);
  _Float16* wqgT  = alloc((size_t)3 * C * C);
  _Float16* wpgT  = alloc((size_t)C * C);
  _Float16* wqlT  = alloc((size_t)3 * C * C);
  _Float16* wplT  = alloc((size_t)C * C);
  _Float16* w1T   = alloc((size_t)C * H);
  _Float16* w2T   = alloc((size_t)H * C);

  auto cdiv = [](size_t a, size_t b) { return (unsigned)((a + b - 1) / b); };

  // Residual stream <- input
  hipMemcpyAsync(xf, x_in, (size_t)TG * C * sizeof(float),
                 hipMemcpyDeviceToDevice, stream);

  // Weight conversions (f32 [K][N] -> f16 [N][K])
  wconv_kernel<<<cdiv((size_t)C * 3 * C, 256), 256, 0, stream>>>(gqkvw, wqgT, C, 3 * C);
  wconv_kernel<<<cdiv((size_t)C * C, 256), 256, 0, stream>>>(gprojw, wpgT, C, C);
  wconv_kernel<<<cdiv((size_t)C * 3 * C, 256), 256, 0, stream>>>(lqkvw, wqlT, C, 3 * C);
  wconv_kernel<<<cdiv((size_t)C * C, 256), 256, 0, stream>>>(lprojw, wplT, C, C);
  wconv_kernel<<<cdiv((size_t)C * H, 256), 256, 0, stream>>>(w1, w1T, C, H);
  wconv_kernel<<<cdiv((size_t)H * C, 256), 256, 0, stream>>>(w2, w2T, H, C);

  const float scale = 0.125f;  // 64^-0.5

  // ---- Global MHA branch ----
  ln_f16_kernel<<<cdiv(TG, 8), 256, 0, stream>>>(xf, ln1g, ln1b, h16, TG, 0, 0, 0);
  gemm_wmma_kernel<<<dim3(cdiv(TG, 128), (3 * C) / 64), 128, 0, stream>>>(
      h16, wqgT, gqkvb, TG, C, 3 * C, 0, qkv16, nullptr, 0, 0);
  qkv_split_kernel<<<cdiv((size_t)BHG * StG * 64, 256), 256, 0, stream>>>(
      qkv16, Qb, Kb, Vb, B, NH, N, StG);
  attn_wmma_kernel<<<dim3(cdiv(StG, 64), NH, B), 128, 0, stream>>>(
      Qb, Kb, Vb, O16, N, StG, scale);
  gemm_wmma_kernel<<<dim3(cdiv(TG, 128), C / 64), 128, 0, stream>>>(
      O16, wpgT, gprojb, TG, C, C, 2, nullptr, xf, 0, 0);

  // ---- Local (per-image) MHA branch ----
  ln_f16_kernel<<<cdiv(TL, 8), 256, 0, stream>>>(xf, ln2g, ln2b, h16, TL, 1,
                                                 N - 1, N);
  gemm_wmma_kernel<<<dim3(cdiv(TL, 128), (3 * C) / 64), 128, 0, stream>>>(
      h16, wqlT, lqkvb, TL, C, 3 * C, 0, qkv16, nullptr, 0, 0);
  qkv_split_kernel<<<cdiv((size_t)BHL * StL * 64, 256), 256, 0, stream>>>(
      qkv16, Qb, Kb, Vb, B * NI, NH, P, StL);
  attn_wmma_kernel<<<dim3(cdiv(StL, 64), NH, B * NI), 128, 0, stream>>>(
      Qb, Kb, Vb, O16, P, StL, scale);
  gemm_wmma_kernel<<<dim3(cdiv(TL, 128), C / 64), 128, 0, stream>>>(
      O16, wplT, lprojb, TL, C, C, 3, nullptr, xf, N - 1, N);

  // ---- MLP ----
  ln_f16_kernel<<<cdiv(TG, 8), 256, 0, stream>>>(xf, ln3g, ln3b, h16, TG, 0, 0, 0);
  gemm_wmma_kernel<<<dim3(cdiv(TG, 128), H / 64), 128, 0, stream>>>(
      h16, w1T, b1, TG, C, H, 1, g16, nullptr, 0, 0);
  gemm_wmma_kernel<<<dim3(cdiv(TG, 128), C / 64), 128, 0, stream>>>(
      g16, w2T, b2, TG, H, C, 2, nullptr, xf, 0, 0);
}
